// SeiturdModel_86904368267500
// MI455X (gfx1250) — compile-verified, matrix operand order checked by reading, not run on hardware
//
#include <hip/hip_runtime.h>
#include <hip/hip_bf16.h>

#define R        4096
#define STEPS    729            // lax.scan over days 1..729
#define NBLK     32
#define NTHR     256
#define KTILES   128            // 4096 / 32
#define WAVES    8              // NTHR / 32

typedef __attribute__((ext_vector_type(16))) __bf16 v16bf;
typedef __attribute__((ext_vector_type(8)))  float  v8f;

union Frag {
    uint4          q[2];
    v16bf          v;
    unsigned short s[16];
};

__device__ __forceinline__ unsigned short f2bf(float f) {
    union { __hip_bfloat16 h; unsigned short u; } c;
    c.h = __float2bfloat16(f);
    return c.u;
}

// ---------------------------------------------------------------------------
// One-shot: convert fp32 adjacency (row-major 4096x4096) into bf16 tiles laid
// out in the exact CDNA5 WMMA A-fragment order, so the hot loop does two
// contiguous global_load_b128 per lane per tile. Tile (rt,kt) = rows rt*16..+15,
// cols kt*32..+31; 1 KB per tile; lane L owns 32 contiguous bytes.
//   lane half = L>>4 ; elems 0..7 -> K = 8*half + e ; elems 8..15 -> K = 16 + 8*half + (e-8)
// Also zeroes the grid-barrier counter.
// ---------------------------------------------------------------------------
__global__ void seiturd_tile_bf16(const float* __restrict__ adj,
                                  unsigned short* __restrict__ adjT,
                                  unsigned* __restrict__ ctr) {
    unsigned gid  = blockIdx.x * blockDim.x + threadIdx.x;
    if (gid == 0) *ctr = 0u;
    unsigned tile = gid >> 5;          // 0 .. 32767  (= 256 rt * 128 kt)
    unsigned lane = gid & 31u;
    unsigned rt   = tile >> 7;
    unsigned kt   = tile & 127u;
    unsigned row  = rt * 16 + (lane & 15u);
    unsigned half = lane >> 4;
    unsigned c0   = kt * 32 + half * 8;   // first run of 8 columns
    unsigned c1   = c0 + 16;              // second run of 8 columns

    const float4* rp = (const float4*)(adj + (size_t)row * R);
    float4 f0 = rp[c0 >> 2];
    float4 f1 = rp[(c0 >> 2) + 1];
    float4 f2 = rp[c1 >> 2];
    float4 f3 = rp[(c1 >> 2) + 1];

    Frag o;
    o.s[0]  = f2bf(f0.x); o.s[1]  = f2bf(f0.y); o.s[2]  = f2bf(f0.z); o.s[3]  = f2bf(f0.w);
    o.s[4]  = f2bf(f1.x); o.s[5]  = f2bf(f1.y); o.s[6]  = f2bf(f1.z); o.s[7]  = f2bf(f1.w);
    o.s[8]  = f2bf(f2.x); o.s[9]  = f2bf(f2.y); o.s[10] = f2bf(f2.z); o.s[11] = f2bf(f2.w);
    o.s[12] = f2bf(f3.x); o.s[13] = f2bf(f3.y); o.s[14] = f2bf(f3.z); o.s[15] = f2bf(f3.w);

    uint4* dst = (uint4*)(adjT + ((size_t)tile << 9)) + lane * 2;  // tile*1024B + lane*32B
    dst[0] = o.q[0];
    dst[1] = o.q[1];
}

__global__ void seiturd_zero_ctr(unsigned* ctr) { *ctr = 0u; }

// ---------------------------------------------------------------------------
// Persistent kernel: 32 blocks x 256 threads. Block b owns rows/regions
// [b*128, b*128+128). Each of its 8 waves computes one 16-row tile of
// y = adjacency @ I_prev per day via v_wmma_f32_16x16x32_bf16 (I broadcast
// across the 16 B-columns). y handoff is intra-block via LDS; the 7-state
// update runs in registers of threads 0..127; one global barrier per day.
// ---------------------------------------------------------------------------
__global__ void __launch_bounds__(NTHR)
seiturd_persistent(const float* __restrict__ adj,
                   const unsigned short* __restrict__ adjT,
                   unsigned* __restrict__ ctr,
                   const float* __restrict__ S0,  const float* __restrict__ E0,
                   const float* __restrict__ I0,  const float* __restrict__ T0,
                   const float* __restrict__ U0,  const float* __restrict__ Rr0,
                   const float* __restrict__ Dd0,
                   const float* __restrict__ lnE, const float* __restrict__ lnI,
                   const float* __restrict__ lnT,
                   const float* __restrict__ det, const float* __restrict__ rec,
                   const float* __restrict__ cI,
                   float* __restrict__ out, int use_wmma) {
    __shared__ __align__(16) unsigned short lds_I[R];   // bf16 bits of I_prev
    __shared__ float lds_y[128];                        // block's matvec rows

    const int tid  = threadIdx.x;
    const int lane = tid & 31;
    const int wave = tid >> 5;
    const int half = lane >> 4;
    const int blk  = blockIdx.x;
    const int rt   = blk * WAVES + wave;                // this wave's row-tile

    const float kE = expf(lnE[0]);
    const float kI = expf(lnI[0]);
    const float kT = expf(lnT[0]);

    // Per-thread persistent state (threads 0..127 own region r).
    const int r = blk * 128 + tid;
    float S = 0.f, E = 0.f, I = 0.f, T = 0.f, U = 0.f, Rr = 0.f, Dd = 0.f;
    if (tid < 128) {
        S = S0[r]; E = E0[r]; I = I0[r]; T = T0[r];
        U = U0[r]; Rr = Rr0[r]; Dd = Dd0[r];
    }

    const uint4* tbase = (const uint4*)(adjT + ((size_t)rt * KTILES << 9));

    for (int t = 0; t < STEPS; ++t) {
        const float* Iprev = (t == 0) ? I0 : (out + ((size_t)(t - 1) * 7 + 2) * R);

        // Stage I_prev into LDS as bf16 (16 elements per thread, no divergence).
        for (int i = tid; i < R; i += NTHR)
            lds_I[i] = f2bf(Iprev[i]);
        __syncthreads();

        if (use_wmma) {
            v8f acc = {0.f, 0.f, 0.f, 0.f, 0.f, 0.f, 0.f, 0.f};
            const uint4* lp = (const uint4*)lds_I;      // 8 bf16 per uint4
#pragma unroll 4
            for (int kt = 0; kt < KTILES; ++kt) {
                Frag a, b;
                const uint4* tp = tbase + (size_t)kt * 64 + lane * 2;
                a.q[0] = tp[0];
                a.q[1] = tp[1];
                // B[k][n] = I[kt*32 + k] broadcast over n (same K-map as A).
                int o0 = kt * 4 + half;                 // (kt*32 + 8*half)/8
                b.q[0] = lp[o0];
                b.q[1] = lp[o0 + 2];                    // +16 bf16
                acc = __builtin_amdgcn_wmma_f32_16x16x32_bf16(
                        false, a.v, false, b.v, (short)0, acc, false, false);
            }
            // D layout: VGPR j -> M=j (lanes 0-15), M=8+j (lanes 16-31); all N equal.
            if (lane == 0) {
#pragma unroll
                for (int j = 0; j < 8; ++j) lds_y[wave * 16 + j] = acc[j];
            } else if (lane == 16) {
#pragma unroll
                for (int j = 0; j < 8; ++j) lds_y[wave * 16 + 8 + j] = acc[j];
            }
        } else {
            // fp32 fallback: lane pairs split K halves of one row.
            const int row = rt * 16 + (lane & 15);
            const float* arow = adj + (size_t)row * R;
            float acc = 0.f;
            const int k0 = half * 2048;
            for (int k = k0; k < k0 + 2048; ++k)
                acc = __builtin_fmaf(arow[k], Iprev[k], acc);
            acc += __shfl_xor(acc, 16, 32);
            if (lane < 16) lds_y[wave * 16 + lane] = acc;
        }
        __syncthreads();

        // Elementwise SEITURD update for this block's 128 regions.
        if (tid < 128) {
            const float  y    = lds_y[tid];
            const size_t pidx = (size_t)(t + 1) * R + r;   // xs = params[1:]
            const float  cIv  = cI[pidx];
            const float  dv   = det[pidx];
            const float  rv   = rec[pidx];

            const float flow_SE = cIv * y * S;
            const float flow_EI = kE * E;
            const float out_I   = kI * I;
            const float out_T   = kT * T;
            S  -= flow_SE;
            E  += flow_SE - flow_EI;
            I  += flow_EI - out_I;
            T  += dv * out_I - out_T;
            U  += (1.f - dv) * out_I;
            Rr += rv * out_T;
            Dd += (1.f - rv) * out_T;

            float* ob = out + (size_t)t * 7 * R + r;
            ob[0 * R] = S;  ob[1 * R] = E;  ob[2 * R] = I;  ob[3 * R] = T;
            ob[4 * R] = U;  ob[5 * R] = Rr; ob[6 * R] = Dd;
        }

        // Grid-wide barrier (monotonic counter; next day reads out[t][2]).
        if (t + 1 < STEPS) {
            __syncthreads();
            if (tid == 0) {
                __threadfence();
                __hip_atomic_fetch_add(ctr, 1u, __ATOMIC_RELEASE, __HIP_MEMORY_SCOPE_AGENT);
                const unsigned target = (unsigned)(t + 1) * (unsigned)gridDim.x;
                while (__hip_atomic_load(ctr, __ATOMIC_ACQUIRE, __HIP_MEMORY_SCOPE_AGENT) < target)
                    __builtin_amdgcn_s_sleep(1);
            }
            __syncthreads();
            __threadfence();
        }
    }
}

extern "C" void kernel_launch(void* const* d_in, const int* in_sizes, int n_in,
                              void* d_out, int out_size, void* d_ws, size_t ws_size,
                              hipStream_t stream) {
    const float* adj = (const float*)d_in[0];
    const float* S0  = (const float*)d_in[1];
    const float* E0  = (const float*)d_in[2];
    const float* I0  = (const float*)d_in[3];
    const float* T0  = (const float*)d_in[4];
    const float* U0  = (const float*)d_in[5];
    const float* R0  = (const float*)d_in[6];
    const float* D0  = (const float*)d_in[7];
    const float* lnE = (const float*)d_in[8];
    const float* lnI = (const float*)d_in[9];
    const float* lnT = (const float*)d_in[10];
    const float* det = (const float*)d_in[11];
    const float* rec = (const float*)d_in[12];
    const float* cI  = (const float*)d_in[13];
    float* out = (float*)d_out;

    unsigned*       ctr  = (unsigned*)d_ws;
    unsigned short* adjT = (unsigned short*)((char*)d_ws + 256);
    const size_t need = 256 + (size_t)R * R * sizeof(unsigned short);   // ~32 MB
    const int use_wmma = (ws_size >= need) ? 1 : 0;

    if (use_wmma) {
        // 256 rt * 128 kt tiles * 32 lanes = 1,048,576 threads
        seiturd_tile_bf16<<<4096, 256, 0, stream>>>(adj, adjT, ctr);
    } else {
        seiturd_zero_ctr<<<1, 1, 0, stream>>>(ctr);
    }
    seiturd_persistent<<<NBLK, NTHR, 0, stream>>>(
        adj, adjT, ctr, S0, E0, I0, T0, U0, R0, D0,
        lnE, lnI, lnT, det, rec, cI, out, use_wmma);
}
